// PaiNNQMEncoder_56349970923948
// MI455X (gfx1250) — compile-verified, matrix operand order checked by reading, not run on hardware
//
#include <hip/hip_runtime.h>
#include <hip/hip_bf16.h>
#include <math.h>

// ---------------------------------------------------------------------------
// Sizes (fixed by the reference)
// ---------------------------------------------------------------------------
#define N_ATOM   8192
#define L_STACK  6
#define F_DIM    256
#define FH_DIM   128
#define D_RBF    32
#define E_EDGE   131072
#define P_PROBE  32768
#define IN_DIM   800      // 32 + 3*256
#define CUTOFF_R 4.0f
#define PI_F     3.14159265358979323846f

#define G_ATOM   8                    // atoms per block in attention kernel
#define A_ROWS   (G_ATOM * L_STACK)   // 48 rows = 3 M-tiles of 16
#define M_EDGE   32                   // edges per block in edge kernel

typedef __bf16 bhalf;
typedef __attribute__((ext_vector_type(16))) bhalf v16bf;
typedef __attribute__((ext_vector_type(8)))  bhalf v8bf;
typedef __attribute__((ext_vector_type(8)))  float v8f;

__device__ __forceinline__ float silu_f(float v) {
    return v / (1.0f + __expf(-v));
}

// ---------------------------------------------------------------------------
// Fragment helper: 16-bit WMMA operand lane layout (A and B identical form)
//   lane = half*16 + r ; row(M)/col(N) = r
//   element i (0..15): k = (i & 7) + 16*(i >> 3) + 8*half
//   -> two contiguous 8-element (16-byte) runs: [8*half, +8) and [16+8*half, +8)
// ---------------------------------------------------------------------------
__device__ __forceinline__ v16bf load_frag(const bhalf* __restrict__ row_ptr, int half) {
    v8bf lo = *(const v8bf*)(row_ptr + 8 * half);
    v8bf hi = *(const v8bf*)(row_ptr + 16 + 8 * half);
    v16bf out;
    #pragma unroll
    for (int i = 0; i < 8; ++i) { out[i] = lo[i]; out[i + 8] = hi[i]; }
    return out;
}

#define WMMA_BF16(A, B, C) \
    __builtin_amdgcn_wmma_f32_16x16x32_bf16(false, (A), false, (B), (short)0, (C), false, false)

// ---------------------------------------------------------------------------
// Kernel Z: zero output (d_out is poisoned by the harness)
// ---------------------------------------------------------------------------
__global__ void zero_rho_kernel(float* __restrict__ rho) {
    int i = blockIdx.x * blockDim.x + threadIdx.x;
    if (i < P_PROBE) rho[i] = 0.0f;
}

// ---------------------------------------------------------------------------
// Kernel W: convert/transpose weights to bf16 so a WMMA B-fragment is two
// contiguous 16-byte runs per lane.
//   w1t [n*800 + k] = msg_w1 [k*256 + n]   (800x256 -> 256x800)
//   w2t [n*256 + k] = msg_w2 [k*128 + n]   (256x128 -> 128x256)
//   aw1t[j*256 + f] = attn_w1[f*128 + j]   (256x128 -> 128x256)
// ---------------------------------------------------------------------------
__global__ void convert_weights_kernel(const float* __restrict__ w1,
                                       const float* __restrict__ w2,
                                       const float* __restrict__ aw1,
                                       bhalf* __restrict__ w1t,
                                       bhalf* __restrict__ w2t,
                                       bhalf* __restrict__ aw1t) {
    int idx = blockIdx.x * blockDim.x + threadIdx.x;
    if (idx < IN_DIM * F_DIM) {
        int k = idx / F_DIM, n = idx % F_DIM;
        w1t[n * IN_DIM + k] = (bhalf)w1[idx];
        return;
    }
    int idx2 = idx - IN_DIM * F_DIM;
    if (idx2 < F_DIM * FH_DIM) {
        int k = idx2 / FH_DIM, n = idx2 % FH_DIM;
        w2t[n * F_DIM + k] = (bhalf)w2[idx2];
        return;
    }
    int idx3 = idx2 - F_DIM * FH_DIM;
    if (idx3 < F_DIM * FH_DIM) {
        int f = idx3 / FH_DIM, j = idx3 % FH_DIM;
        aw1t[j * F_DIM + f] = (bhalf)aw1[idx3];
    }
}

// ---------------------------------------------------------------------------
// Kernel A: per-atom attention with WMMA. 8 atoms (48 rows) per block,
// 256 threads = 8 waves. Produces V_JK (f32), S_JK (bf16), n_i (bf16).
// ---------------------------------------------------------------------------
__global__ __launch_bounds__(256)
void atom_attn_kernel(const float* __restrict__ s_stack,
                      const float* __restrict__ v_stack,
                      const bhalf* __restrict__ aw1t,
                      const float* __restrict__ attn_b1,
                      const float* __restrict__ attn_w2,
                      const float* __restrict__ attn_b2,
                      float* __restrict__ VJK,
                      bhalf* __restrict__ SJKb,
                      bhalf* __restrict__ NIb) {
    const int t   = threadIdx.x;
    const int na0 = blockIdx.x * G_ATOM;

    __shared__ __align__(16) bhalf sA[A_ROWS * F_DIM];   // 24 KB
    __shared__ float logit_sh[A_ROWS];
    __shared__ float a_sh[A_ROWS];

    // stage s rows as bf16 A-tile; init logit accumulators with attn_b2
    const float* sbase = s_stack + (size_t)na0 * L_STACK * F_DIM;
    for (int i = t; i < A_ROWS * F_DIM; i += 256) sA[i] = (bhalf)sbase[i];
    if (t < A_ROWS) logit_sh[t] = attn_b2[0];
    __syncthreads();

    const int wave = t >> 5, lane = t & 31;
    const int half = lane >> 4, r = lane & 15;
    const int j = wave * 16 + r;          // attention hidden column, 0..127

    // h = s @ attn_w1 : 3 M-tiles x 1 N-tile per wave, K = 256 = 8 chunks
    v8f h0 = {}; v8f h1 = {}; v8f h2 = {};
    for (int kc = 0; kc < F_DIM / 32; ++kc) {
        const int kb = kc * 32;
        v16bf bfr = load_frag(aw1t + (size_t)j * F_DIM + kb, half);
        v16bf a0  = load_frag(&sA[(0 * 16 + r) * F_DIM + kb], half);
        v16bf a1  = load_frag(&sA[(1 * 16 + r) * F_DIM + kb], half);
        v16bf a2  = load_frag(&sA[(2 * 16 + r) * F_DIM + kb], half);
        h0 = WMMA_BF16(a0, bfr, h0);
        h1 = WMMA_BF16(a1, bfr, h1);
        h2 = WMMA_BF16(a2, bfr, h2);
    }
    // logits[row] += silu(h[row,j] + b1[j]) * attn_w2[j]  (LDS f32 atomics)
    {
        const float bias = attn_b1[j];
        const float w2j  = attn_w2[j];
        #pragma unroll
        for (int jj = 0; jj < 8; ++jj) {
            int row = jj + 8 * half;
            atomicAdd(&logit_sh[row],      silu_f(h0[jj] + bias) * w2j);
            atomicAdd(&logit_sh[16 + row], silu_f(h1[jj] + bias) * w2j);
            atomicAdd(&logit_sh[32 + row], silu_f(h2[jj] + bias) * w2j);
        }
    }
    __syncthreads();

    // softmax over L per atom
    if (t < G_ATOM) {
        const float* lg = &logit_sh[t * L_STACK];
        float mx = lg[0];
        #pragma unroll
        for (int l = 1; l < L_STACK; ++l) mx = fmaxf(mx, lg[l]);
        float e[L_STACK], s = 0.f;
        #pragma unroll
        for (int l = 0; l < L_STACK; ++l) { e[l] = __expf(lg[l] - mx); s += e[l]; }
        #pragma unroll
        for (int l = 0; l < L_STACK; ++l) a_sh[t * L_STACK + l] = e[l] / s;
    }
    __syncthreads();

    // aggregation (memory-bound): thread t owns feature f = t, loop atoms
    const int f = t;
    for (int g = 0; g < G_ATOM; ++g) {
        const int n = na0 + g;
        const float* aw = &a_sh[g * L_STACK];

        const float* sp = s_stack + (size_t)n * L_STACK * F_DIM;
        float acc = 0.f;
        #pragma unroll
        for (int l = 0; l < L_STACK; ++l) acc += aw[l] * sp[l * F_DIM + f];
        SJKb[(size_t)n * F_DIM + f] = (bhalf)acc;

        const float* vp = v_stack + (size_t)n * L_STACK * 3 * F_DIM;
        float nsum = 0.f;
        #pragma unroll
        for (int c = 0; c < 3; ++c) {
            float vc = 0.f;
            #pragma unroll
            for (int l = 0; l < L_STACK; ++l)
                vc += aw[l] * vp[(l * 3 + c) * F_DIM + f];
            VJK[((size_t)n * 3 + c) * F_DIM + f] = vc;
            nsum += vc * vc;
        }
        NIb[(size_t)n * F_DIM + f] = (bhalf)sqrtf(nsum);
    }
}

// ---------------------------------------------------------------------------
// Kernel C: edge message MLP with bf16 WMMA. 32 edges per workgroup,
// 256 threads = 8 waves; each wave computes a 2x2 tile grid in layer 1
// (B fragments reused across 2 M-tiles -> halves L2 weight traffic).
// Atile is reused for the layer-1 activations after a barrier, so LDS
// stays ~52 KB.
// ---------------------------------------------------------------------------
__global__ __launch_bounds__(256)
void edge_mlp_kernel(const int*   __restrict__ probe_edges,
                     const float* __restrict__ disp,
                     const float* __restrict__ cell,
                     const float* __restrict__ atom_xyz,
                     const float* __restrict__ probe_xyz,
                     const float* __restrict__ VJK,
                     const bhalf* __restrict__ SJKb,
                     const bhalf* __restrict__ NIb,
                     const bhalf* __restrict__ w1t,
                     const float* __restrict__ b1,
                     const bhalf* __restrict__ w2t,
                     const float* __restrict__ b2,
                     const float* __restrict__ w3,
                     const float* __restrict__ b3,
                     float* __restrict__ rho) {
    const int t  = threadIdx.x;
    const int e0 = blockIdx.x * M_EDGE;

    __shared__ __align__(16) bhalf Atile[M_EDGE * IN_DIM];  // 51.2 KB (reused as act1)
    __shared__ float rh[3][M_EDGE];
    __shared__ float cwA[M_EDGE], distA[M_EDGE];
    __shared__ int   aidx[M_EDGE], pidx[M_EDGE];
    __shared__ float m3acc[M_EDGE];

    // ---- Phase 0: per-edge geometry --------------------------------------
    if (t < M_EDGE) {
        const int e = e0 + t;
        const int a = probe_edges[2 * e];
        const int p = probe_edges[2 * e + 1];
        aidx[t] = a; pidx[t] = p;
        float d0 = disp[3 * e], d1 = disp[3 * e + 1], d2 = disp[3 * e + 2];
        float sx = d0 * cell[0] + d1 * cell[3] + d2 * cell[6];
        float sy = d0 * cell[1] + d1 * cell[4] + d2 * cell[7];
        float sz = d0 * cell[2] + d1 * cell[5] + d2 * cell[8];
        float dx = probe_xyz[3 * p]     - (atom_xyz[3 * a]     + sx);
        float dy = probe_xyz[3 * p + 1] - (atom_xyz[3 * a + 1] + sy);
        float dz = probe_xyz[3 * p + 2] - (atom_xyz[3 * a + 2] + sz);
        float dist = sqrtf(dx * dx + dy * dy + dz * dz);
        float inv  = 1.0f / (dist + 1e-8f);
        rh[0][t] = dx * inv; rh[1][t] = dy * inv; rh[2][t] = dz * inv;
        distA[t] = dist;
        cwA[t] = (dist < CUTOFF_R) ? 0.5f * (__cosf(PI_F * dist / CUTOFF_R) + 1.0f) : 0.0f;
    }
    __syncthreads();

    // ---- Phase 1: e_pi (cols 0..31) --------------------------------------
    for (int idx = t; idx < M_EDGE * D_RBF; idx += 256) {
        int e = idx >> 5, k = idx & 31;
        float d = distA[e];
        float val = __sinf(d * (float)(k + 1) * (PI_F / CUTOFF_R)) / (d + 1e-8f);
        Atile[e * IN_DIM + k] = (bhalf)val;
    }
    // ---- Phase 2: q_pi (32..287), n_i (288..543), S_JK (544..799) --------
    for (int idx = t; idx < M_EDGE * F_DIM; idx += 256) {
        int e = idx >> 8, f = idx & 255;
        const size_t ab = (size_t)aidx[e];
        const float* vb = VJK + ab * 3 * F_DIM + f;
        float q = vb[0] * rh[0][e] + vb[F_DIM] * rh[1][e] + vb[2 * F_DIM] * rh[2][e];
        Atile[e * IN_DIM +  32 + f] = (bhalf)q;
        Atile[e * IN_DIM + 288 + f] = NIb [ab * F_DIM + f];
        Atile[e * IN_DIM + 544 + f] = SJKb[ab * F_DIM + f];
    }
    __syncthreads();

    const int wave = t >> 5, lane = t & 31;
    const int half = lane >> 4, r = lane & 15;

    // ---- Layer 1: (32 x 800) @ (800 x 256), 2 M-tiles x 2 N-tiles/wave ---
    v8f c00 = {}; v8f c01 = {}; v8f c10 = {}; v8f c11 = {};
    const int n0 = wave * 32 + r;
    const int n1 = n0 + 16;
    for (int kc = 0; kc < IN_DIM / 32; ++kc) {
        const int kb = kc * 32;
        v16bf a0  = load_frag(&Atile[r * IN_DIM + kb], half);
        v16bf a1  = load_frag(&Atile[(16 + r) * IN_DIM + kb], half);
        v16bf bf0 = load_frag(w1t + (size_t)n0 * IN_DIM + kb, half);
        v16bf bf1 = load_frag(w1t + (size_t)n1 * IN_DIM + kb, half);
        c00 = WMMA_BF16(a0, bf0, c00);
        c10 = WMMA_BF16(a1, bf0, c10);
        c01 = WMMA_BF16(a0, bf1, c01);
        c11 = WMMA_BF16(a1, bf1, c11);
    }
    __syncthreads();   // everyone done reading Atile; accumulators live in regs

    // bias + SiLU -> act (bf16), reusing Atile with stride 256
    {
        bhalf* act = Atile;
        const float bias0 = b1[n0], bias1 = b1[n1];
        #pragma unroll
        for (int jj = 0; jj < 8; ++jj) {
            const int m0 = jj + 8 * half;
            const int m1 = 16 + m0;
            act[m0 * F_DIM + n0] = (bhalf)silu_f(c00[jj] + bias0);
            act[m0 * F_DIM + n1] = (bhalf)silu_f(c01[jj] + bias1);
            act[m1 * F_DIM + n0] = (bhalf)silu_f(c10[jj] + bias0);
            act[m1 * F_DIM + n1] = (bhalf)silu_f(c11[jj] + bias1);
        }
        if (t < M_EDGE) m3acc[t] = b3[0];
    }
    __syncthreads();

    // ---- Layer 2: (32 x 256) @ (256 x 128), 2 M-tiles x 1 N-tile/wave ----
    v8f d0 = {}; v8f d1 = {};
    const int n2 = wave * 16 + r;     // 0..127
    {
        const bhalf* act = Atile;
        for (int kc = 0; kc < F_DIM / 32; ++kc) {
            const int kb = kc * 32;
            v16bf a0  = load_frag(&act[r * F_DIM + kb], half);
            v16bf a1  = load_frag(&act[(16 + r) * F_DIM + kb], half);
            v16bf bfr = load_frag(w2t + (size_t)n2 * F_DIM + kb, half);
            d0 = WMMA_BF16(a0, bfr, d0);
            d1 = WMMA_BF16(a1, bfr, d1);
        }
    }
    // ---- Layer 3: silu, dot with w3 via LDS f32 atomics ------------------
    {
        const float bias2 = b2[n2];
        const float w3n   = w3[n2];
        #pragma unroll
        for (int jj = 0; jj < 8; ++jj) {
            const int m0 = jj + 8 * half;
            atomicAdd(&m3acc[m0],      silu_f(d0[jj] + bias2) * w3n);
            atomicAdd(&m3acc[16 + m0], silu_f(d1[jj] + bias2) * w3n);
        }
    }
    __syncthreads();

    // ---- cutoff + scatter into rho ---------------------------------------
    if (t < M_EDGE) {
        atomicAdd(&rho[pidx[t]], m3acc[t] * cwA[t]);
    }
}

// ---------------------------------------------------------------------------
// Launcher
// ---------------------------------------------------------------------------
extern "C" void kernel_launch(void* const* d_in, const int* in_sizes, int n_in,
                              void* d_out, int out_size, void* d_ws, size_t ws_size,
                              hipStream_t stream) {
    const float* s_stack   = (const float*)d_in[0];
    const float* v_stack   = (const float*)d_in[1];
    const float* atom_xyz  = (const float*)d_in[2];
    const float* probe_xyz = (const float*)d_in[3];
    const float* cell      = (const float*)d_in[4];
    const float* disp      = (const float*)d_in[5];
    const float* attn_w1   = (const float*)d_in[6];
    const float* attn_b1   = (const float*)d_in[7];
    const float* attn_w2   = (const float*)d_in[8];
    const float* attn_b2   = (const float*)d_in[9];
    const float* msg_w1    = (const float*)d_in[10];
    const float* msg_b1    = (const float*)d_in[11];
    const float* msg_w2    = (const float*)d_in[12];
    const float* msg_b2    = (const float*)d_in[13];
    const float* msg_w3    = (const float*)d_in[14];
    const float* msg_b3    = (const float*)d_in[15];
    const int*   probe_edges = (const int*)d_in[16];

    float* rho = (float*)d_out;

    // Workspace layout
    char* ws = (char*)d_ws;
    size_t off = 0;
    float* VJK  = (float*)(ws + off); off += (size_t)N_ATOM * 3 * F_DIM * 4;  // 25,165,824
    bhalf* SJKb = (bhalf*)(ws + off); off += (size_t)N_ATOM * F_DIM * 2;      //  4,194,304
    bhalf* NIb  = (bhalf*)(ws + off); off += (size_t)N_ATOM * F_DIM * 2;      //  4,194,304
    bhalf* w1t  = (bhalf*)(ws + off); off += (size_t)IN_DIM * F_DIM * 2;      //    409,600
    bhalf* w2t  = (bhalf*)(ws + off); off += (size_t)F_DIM * FH_DIM * 2;      //     65,536
    bhalf* aw1t = (bhalf*)(ws + off); off += (size_t)F_DIM * FH_DIM * 2;      //     65,536

    zero_rho_kernel<<<(P_PROBE + 255) / 256, 256, 0, stream>>>(rho);

    int conv_elems = IN_DIM * F_DIM + 2 * F_DIM * FH_DIM;
    convert_weights_kernel<<<(conv_elems + 255) / 256, 256, 0, stream>>>(
        msg_w1, msg_w2, attn_w1, w1t, w2t, aw1t);

    atom_attn_kernel<<<N_ATOM / G_ATOM, 256, 0, stream>>>(
        s_stack, v_stack, aw1t, attn_b1, attn_w2, attn_b2, VJK, SJKb, NIb);

    edge_mlp_kernel<<<E_EDGE / M_EDGE, 256, 0, stream>>>(
        probe_edges, disp, cell, atom_xyz, probe_xyz,
        VJK, SJKb, NIb, w1t, msg_b1, w2t, msg_b2, msg_w3, msg_b3, rho);
}
